// DeeperGCN_G_85950885527884
// MI455X (gfx1250) — matrix operand order repaired and checked by reading, not compiled, not run on hardware
//
#include <hip/hip_runtime.h>

typedef __attribute__((ext_vector_type(16))) _Float16     v16h;
typedef __attribute__((ext_vector_type(8)))  float        v8f;
typedef __attribute__((ext_vector_type(4)))  unsigned int v4u;
typedef __attribute__((ext_vector_type(8)))  int          v8i;
typedef __attribute__((ext_vector_type(4)))  int          v4i;

#define H 64
#define MSG_EPS 1e-7f
#define NEG_INF_FLIPPED 0x807fffff   // float_flip(-inf)

// ---- monotone float<->int mapping (for atomicMax on floats) --------------
__device__ __forceinline__ int float_flip(float f) {
  int i = __float_as_int(f);
  return (i < 0) ? (i ^ 0x7fffffff) : i;   // involution, order-preserving
}
__device__ __forceinline__ float float_unflip(int i) {
  int j = (i < 0) ? (i ^ 0x7fffffff) : i;
  return __int_as_float(j);
}

// ---- Tensor Data Mover: async 2D tile (16 rows x 32 f32) -> LDS ----------
__device__ __forceinline__ void tdm_issue_tile(const float* A, int M, int K,
                                               int row0, int k0,
                                               unsigned lds_byte) {
  unsigned long long ga =
      (unsigned long long)(size_t)(A + (size_t)row0 * K + k0);
  unsigned td0 = (unsigned)(K - k0);   // tensor_dim0 (elems remaining in row)
  unsigned td1 = (unsigned)(M - row0); // tensor_dim1 (rows remaining; OOB->0)
  // D# group0: count=1 | lds_addr | global_addr(57b) | type=2
  v4u g0 = {1u, lds_byte, (unsigned)ga,
            (unsigned)((ga >> 32) & 0x01FFFFFFull) | (2u << 30)};
  // D# group1: mask=0, data_size=4B, dims, tile 32x16, dim0 stride = K
  v8i g1 = {(int)(2u << 16),
            (int)((td0 & 0xFFFFu) << 16),
            (int)((td0 >> 16) | ((td1 & 0xFFFFu) << 16)),
            (int)((td1 >> 16) | (32u << 16)),   // tile_dim0 = 32
            16,                                 // tile_dim1 = 16
            (int)(unsigned)K,                   // tensor_dim0_stride
            0, 0};
  v4i z4 = {0, 0, 0, 0};
#if __clang_major__ >= 23
  v8i z8 = {0, 0, 0, 0, 0, 0, 0, 0};
  __builtin_amdgcn_tensor_load_to_lds(g0, g1, z4, z4, z8, 0);
#else
  __builtin_amdgcn_tensor_load_to_lds(g0, g1, z4, z4, 0);
#endif
}

// ============================ WMMA GEMM ===================================
// Y[M,NC] = A[M,K] @ W[K,NC] + bias.  One wave per 16x16 C tile,
// blockDim = 32*(NC/16), grid.x = ceil(M/16).
// A tiles DMA'd into double-buffered LDS by the Tensor Data Mover (wave 0),
// pipelined one tile ahead; W staged transposed in LDS as f16 so B fragments
// are contiguous 32B ds_load reads.
template <int K, int NC>
__global__ void gemm_wmma_kernel(const float* __restrict__ A,
                                 const float* __restrict__ W,
                                 const float* __restrict__ bias,
                                 float* __restrict__ Y, int M) {
  constexpr int NSTEP = K / 32;
  __shared__ __attribute__((aligned(32))) _Float16 sWt[NC * K];   // W^T: [n][k]
  __shared__ __attribute__((aligned(32))) float    sA[2][16 * 32];

  const int lane = threadIdx.x & 31;
  const int wave = threadIdx.x >> 5;
  const int row0 = blockIdx.x * 16;
  const int col0 = wave * 16;
  const int half = lane >> 4;   // lane group 0/1
  const int mr   = lane & 15;

  // kick off the first A-tile DMA; it overlaps the W staging below
  if (threadIdx.x < 32)
    tdm_issue_tile(A, M, K, row0, 0, (unsigned)(size_t)(&sA[0][0]));

  // stage W transposed (coalesced global read, one-time)
  for (int i = threadIdx.x; i < K * NC; i += blockDim.x) {
    int k = i / NC, n = i - k * NC;
    sWt[n * K + k] = (_Float16)W[i];
  }

  v8f c = {};
#pragma unroll
  for (int i = 0; i < NSTEP; ++i) {
    const int k0 = i * 32;
    __syncthreads();  // iter i-1 consumers done (incl. buffer (i+1)&1); sWt staged
    if (threadIdx.x < 32) {
      if (i + 1 < NSTEP) {  // prefetch next tile, then wait for oldest
        tdm_issue_tile(A, M, K, row0, k0 + 32,
                       (unsigned)(size_t)(&sA[(i + 1) & 1][0]));
        __builtin_amdgcn_s_wait_tensorcnt(1);
      } else {
        __builtin_amdgcn_s_wait_tensorcnt(0);
      }
    }
    __syncthreads();  // sA[i&1] ready for all waves

    const float* At = sA[i & 1];
    // A fragment: lane holds M=mr; K contiguous [half*16, +16)
    v16h a;
#pragma unroll
    for (int e = 0; e < 8; ++e) {
      a[e]     = (_Float16)At[(mr << 5) + (half << 4) + e];
      a[e + 8] = (_Float16)At[(mr << 5) + (half << 4) + 8 + e];
    }
    // B fragment: lane = column (col0+mr), elements = consecutive K
    v16h b = *(const v16h*)(sWt + (col0 + mr) * K + k0 + (half << 4));
    c = __builtin_amdgcn_wmma_f32_16x16x32_f16(false, a, false, b,
                                               (short)0, c, false, false);
  }

  // C/D layout: lanes0-15: N=lane, M=v; lanes16-31: N=lane-16, M=v+8
  const float bb = bias[col0 + mr];
  if (row0 + 16 <= M) {         // fast path: full tile, unguarded stores
#pragma unroll
    for (int v = 0; v < 8; ++v)
      Y[(row0 + v + (half << 3)) * NC + col0 + mr] = c[v] + bb;
  } else {
#pragma unroll
    for (int v = 0; v < 8; ++v) {
      int r = row0 + v + (half << 3);
      if (r < M) Y[r * NC + col0 + mr] = c[v] + bb;
    }
  }
}

// ============================ edge passes =================================
__global__ void init_conv_buffers(int* __restrict__ nmax, float* __restrict__ nsum,
                                  float* __restrict__ agg, int n) {
  int i = blockIdx.x * blockDim.x + threadIdx.x;
  if (i < n) { nmax[i] = NEG_INF_FLIPPED; nsum[i] = 0.f; agg[i] = 0.f; }
}

__global__ void edge_max_kernel(const float* __restrict__ x, const int* __restrict__ src,
                                const int* __restrict__ dst, const float* __restrict__ tptr,
                                int* __restrict__ nmax, int E) {
  int tid = blockIdx.x * blockDim.x + threadIdx.x;
  if (tid >= E * 16) return;
  int e = tid >> 4, g = (tid & 15) << 2;
  int s = src[e], d = dst[e];
  float t = tptr[0];
  const float4 xv = *(const float4*)(x + s * H + g);
  float vv[4] = {xv.x, xv.y, xv.z, xv.w};
#pragma unroll
  for (int j = 0; j < 4; ++j) {
    float m = fmaxf(vv[j], 0.f) + MSG_EPS;
    atomicMax(nmax + d * H + g + j, float_flip(m * t));
  }
}

__global__ void finalize_max_kernel(int* __restrict__ buf, int n) {
  int i = blockIdx.x * blockDim.x + threadIdx.x;
  if (i >= n) return;
  float f = float_unflip(buf[i]);
  if (!__builtin_isfinite(f)) f = 0.f;   // isolated nodes -> 0 (matches reference)
  ((float*)buf)[i] = f;
}

__global__ void edge_expsum_kernel(const float* __restrict__ x, const int* __restrict__ src,
                                   const int* __restrict__ dst, const float* __restrict__ tptr,
                                   const float* __restrict__ nmax, float* __restrict__ nsum, int E) {
  int tid = blockIdx.x * blockDim.x + threadIdx.x;
  if (tid >= E * 16) return;
  int e = tid >> 4, g = (tid & 15) << 2;
  int s = src[e], d = dst[e];
  float t = tptr[0];
  const float4 xv = *(const float4*)(x + s * H + g);
  float vv[4] = {xv.x, xv.y, xv.z, xv.w};
#pragma unroll
  for (int j = 0; j < 4; ++j) {
    int idx = d * H + g + j;
    float m = fmaxf(vv[j], 0.f) + MSG_EPS;
    float ex = __expf(m * t - nmax[idx]);
    atomicAdd(nsum + idx, ex);
  }
}

__global__ void edge_weighted_kernel(const float* __restrict__ x, const int* __restrict__ src,
                                     const int* __restrict__ dst, const float* __restrict__ tptr,
                                     const float* __restrict__ nmax, const float* __restrict__ nsum,
                                     float* __restrict__ agg, int E) {
  int tid = blockIdx.x * blockDim.x + threadIdx.x;
  if (tid >= E * 16) return;
  int e = tid >> 4, g = (tid & 15) << 2;
  int s = src[e], d = dst[e];
  float t = tptr[0];
  const float4 xv = *(const float4*)(x + s * H + g);
  float vv[4] = {xv.x, xv.y, xv.z, xv.w};
#pragma unroll
  for (int j = 0; j < 4; ++j) {
    int idx = d * H + g + j;
    float m = fmaxf(vv[j], 0.f) + MSG_EPS;
    float ex = __expf(m * t - nmax[idx]);
    float alpha = ex / (nsum[idx] + 1e-16f);
    atomicAdd(agg + idx, m * alpha);
  }
}

__global__ void add_kernel(float* __restrict__ a, const float* __restrict__ b, int n) {
  int i = blockIdx.x * blockDim.x + threadIdx.x;
  if (i < n) a[i] += b[i];
}

// ============================ LN / concat / head ==========================
// One wave per row; blockDim = 256 (8 rows/block). C in {64,128}.
__global__ void ln_kernel(const float* __restrict__ X, const float* __restrict__ g,
                          const float* __restrict__ b, float* __restrict__ Y,
                          int C, int doRelu, int N) {
  int row = blockIdx.x * (blockDim.x >> 5) + (threadIdx.x >> 5);
  int lane = threadIdx.x & 31;
  if (row >= N) return;
  int per = C >> 5;  // 2 or 4
  float v[4];
  float s = 0.f, s2 = 0.f;
  for (int i = 0; i < per; ++i) {
    float xv = X[row * C + lane + (i << 5)];
    v[i] = xv; s += xv; s2 += xv * xv;
  }
#pragma unroll
  for (int off = 16; off > 0; off >>= 1) {
    s  += __shfl_xor(s, off, 32);
    s2 += __shfl_xor(s2, off, 32);
  }
  float inv = 1.f / (float)C;
  float mu = s * inv;
  float var = s2 * inv - mu * mu;
  float rs = rsqrtf(var + 1e-5f);
  for (int i = 0; i < per; ++i) {
    int cc = lane + (i << 5);
    float o = (v[i] - mu) * rs * g[cc] + b[cc];
    if (doRelu) o = fmaxf(o, 0.f);
    Y[row * C + cc] = o;
  }
}

__global__ void concat_kernel(const float* __restrict__ a, const float* __restrict__ b,
                              float* __restrict__ out, int total) {
  int i = blockIdx.x * blockDim.x + threadIdx.x;
  if (i >= total) return;
  int n = i >> 7, c = i & 127;
  out[i] = (c < H) ? a[n * H + c] : b[n * H + (c - H)];
}

// y[n] = dot(X[n, 0:128], w) + b0.  One wave per row.
__global__ void head_kernel(const float* __restrict__ X, const float* __restrict__ w,
                            const float* __restrict__ bptr, float* __restrict__ out, int N) {
  int row = blockIdx.x * (blockDim.x >> 5) + (threadIdx.x >> 5);
  int lane = threadIdx.x & 31;
  if (row >= N) return;
  float s = 0.f;
#pragma unroll
  for (int i = 0; i < 4; ++i) {
    int c = lane + (i << 5);
    s += X[row * 128 + c] * w[c];
  }
#pragma unroll
  for (int off = 16; off > 0; off >>= 1) s += __shfl_xor(s, off, 32);
  if (lane == 0) out[row] = s + bptr[0];
}

// ============================ host orchestration ==========================
static void run_gen_conv(const float* xin, float* xout,
                         const float* W1, const float* b1, const float* g1, const float* be1,
                         const float* W2, const float* b2,
                         const int* src, const int* dst, const float* tptr,
                         float* nmax, float* nsum, float* agg, float* tmp,
                         int N, int E, hipStream_t stream) {
  const int NH = N * H;
  const int et = E * 16;
  init_conv_buffers<<<(NH + 255) / 256, 256, 0, stream>>>((int*)nmax, nsum, agg, NH);
  edge_max_kernel<<<(et + 255) / 256, 256, 0, stream>>>(xin, src, dst, tptr, (int*)nmax, E);
  finalize_max_kernel<<<(NH + 255) / 256, 256, 0, stream>>>((int*)nmax, NH);
  edge_expsum_kernel<<<(et + 255) / 256, 256, 0, stream>>>(xin, src, dst, tptr, nmax, nsum, E);
  edge_weighted_kernel<<<(et + 255) / 256, 256, 0, stream>>>(xin, src, dst, tptr, nmax, nsum, agg, E);
  add_kernel<<<(NH + 255) / 256, 256, 0, stream>>>(agg, xin, NH);  // out = agg + x
  // MLP: Linear(64->128) -> LN -> ReLU -> Linear(128->64)
  gemm_wmma_kernel<H, 2 * H><<<(N + 15) / 16, 32 * (2 * H / 16), 0, stream>>>(agg, W1, b1, tmp, N);
  ln_kernel<<<(N + 7) / 8, 256, 0, stream>>>(tmp, g1, be1, tmp, 2 * H, 1, N);
  gemm_wmma_kernel<2 * H, H><<<(N + 15) / 16, 32 * (H / 16), 0, stream>>>(tmp, W2, b2, xout, N);
}

extern "C" void kernel_launch(void* const* d_in, const int* in_sizes, int n_in,
                              void* d_out, int out_size, void* d_ws, size_t ws_size,
                              hipStream_t stream) {
  const float* x    = (const float*)d_in[0];
  const int*   ei   = (const int*)d_in[1];
  const float* encW = (const float*)d_in[2];
  const float* encb = (const float*)d_in[3];
  const float* tptr = (const float*)d_in[4];
  const float* W1   = (const float*)d_in[5];
  const float* b1   = (const float*)d_in[6];
  const float* g1   = (const float*)d_in[7];
  const float* be1  = (const float*)d_in[8];
  const float* W2   = (const float*)d_in[9];
  const float* b2   = (const float*)d_in[10];
  const float* ln1g = (const float*)d_in[11];
  const float* ln1b = (const float*)d_in[12];
  const float* ng   = (const float*)d_in[13];
  const float* nb   = (const float*)d_in[14];
  const float* linW = (const float*)d_in[15];
  const float* linb = (const float*)d_in[16];

  const int FIN = 128;
  const int N = in_sizes[0] / FIN;
  const int E = in_sizes[1] / 2;
  const int* src = ei;
  const int* dst = ei + E;

  float* ws = (float*)d_ws;
  const size_t NH = (size_t)N * H;
  float* p_x0   = ws;               // encoder out        [N,64]
  float* p_x1   = p_x0 + NH;        // conv1 out          [N,64]
  float* p_h    = p_x1 + NH;        // conv2 out / relu-ln[N,64]
  float* p_nmax = p_h + NH;         // seg max            [N,64]
  float* p_nsum = p_nmax + NH;      // seg sum            [N,64]
  float* p_agg  = p_nsum + NH;      // aggregate/out      [N,64]
  float* p_tmp  = p_agg + NH;       // MLP hidden         [N,128]
  float* p_cat  = p_nmax;           // reuse nmax+nsum as [N,128] after convs

  // node encoder: x0 = x @ enc_W + enc_b
  gemm_wmma_kernel<128, H><<<(N + 15) / 16, 32 * (H / 16), 0, stream>>>(x, encW, encb, p_x0, N);

  run_gen_conv(p_x0, p_x1, W1, b1, g1, be1, W2, b2, src, dst, tptr,
               p_nmax, p_nsum, p_agg, p_tmp, N, E, stream);
  run_gen_conv(p_x1, p_h,  W1, b1, g1, be1, W2, b2, src, dst, tptr,
               p_nmax, p_nsum, p_agg, p_tmp, N, E, stream);

  // h = relu(LN(h));  x = relu(LN(concat[x1, h]));  out = x @ lin_W + lin_b
  ln_kernel<<<(N + 7) / 8, 256, 0, stream>>>(p_h, ln1g, ln1b, p_h, H, 1, N);
  concat_kernel<<<((N * 128) + 255) / 256, 256, 0, stream>>>(p_x1, p_h, p_cat, N * 128);
  ln_kernel<<<(N + 7) / 8, 256, 0, stream>>>(p_cat, ng, nb, p_cat, 128, 1, N);
  head_kernel<<<(N + 7) / 8, 256, 0, stream>>>(p_cat, linW, linb, (float*)d_out, N);
}
